// ATTBase_84851373900347
// MI455X (gfx1250) — matrix operand order repaired and checked by reading, not compiled, not run on hardware
//
#include <hip/hip_runtime.h>
#include <math.h>

typedef _Float16 v8h  __attribute__((ext_vector_type(8)));
typedef _Float16 v16h __attribute__((ext_vector_type(16)));
typedef float    v8f  __attribute__((ext_vector_type(8)));

#define HN 100        // hidden width
#define NT 7          // N tiles (7*16 = 112 >= 100)
#define AP 384        // activation LDS pitch in halves (3 segments of 128)

// ---- packed weight offsets inside d_ws, in halves ----
#define OFF_A1    0        // actor_w1      32x112
#define OFF_A2    3584     // actor_w2     128x112
#define OFF_AGC   17920    // enc_agent_corr 128x112
#define OFF_LDC   32256    // enc_land_corr  128x112
#define OFF_EFC   46592    // enc_fc_w       128x112
#define OFF_EOUT  60928    // enc_out_w      384x112 (3 segments of 128 <- 100 rows)
#define OFF_CORR  103936   // corr_mat       128x112
#define OFF_FC    118272   // fc_w           128x112
#define OFF_CR1   132608   // crit_w1        256x112 (2 segments)
#define WS_HALVES 161280

// -------------------------------------------------------------------------
// Pack a (Korig x 100) row-major f32 weight into WMMA-B fragment order, f16.
// Fragment order: [(kt*NT + nt)*512 + lane*16 + h]
//   n = (lane&15) + nt*16 ; k = kt*32 + (lane>>4)*16 + h
// seg==1: k maps through 128-aligned segments -> src row = (k>>7)*100 + (k&127)
// -------------------------------------------------------------------------
__global__ void pack_w(const float* __restrict__ W, _Float16* __restrict__ dst,
                       int Kp, int Korig, int seg)
{
    int total = Kp * 112;
    int p = blockIdx.x * blockDim.x + threadIdx.x;
    if (p >= total) return;
    int frag = p >> 9;           // /512
    int r    = p & 511;
    int lane = r >> 4, h = r & 15;
    int kt = frag / NT, nt = frag % NT;
    int n = (lane & 15) + nt * 16;
    int k = kt * 32 + ((lane >> 4) << 4) + h;
    float val = 0.f;
    int srcK; bool ok;
    if (seg) { int off = k & 127; srcK = (k >> 7) * HN + off; ok = (off < HN) && (srcK < Korig); }
    else     { srcK = k; ok = (k < Korig); }
    if (ok && n < HN) val = W[(size_t)srcK * HN + n];
    dst[p] = (_Float16)val;
}

// ---- branch-free tanh: exact identity on clamped input --------------------
__device__ __forceinline__ float fast_tanh(float x)
{
    float cx = fminf(fmaxf(x, -15.f), 15.f);
    float t  = __expf(2.f * cx);
    return 1.f - 2.f / (t + 1.f);
}

// ---- fragment loaders -----------------------------------------------------
// A (16x32 f16): lanes 0-15 row=lane hold K=kb..kb+15 (kb=0);
// lanes 16-31 same rows with kb=16.
__device__ __forceinline__ v16h ldsA(const _Float16* act, int pitch, int lane, int kbase)
{
    int row = lane & 15, kb = (lane >> 4) * 16;
    v8h lo = *(const v8h*)(act + row * pitch + kbase + kb);
    v8h hi = *(const v8h*)(act + row * pitch + kbase + kb + 8);
    return __builtin_shufflevector(lo, hi, 0,1,2,3,4,5,6,7,8,9,10,11,12,13,14,15);
}
// B (32x16 f16) pre-packed: lane reads 16 contiguous halves.
__device__ __forceinline__ v16h ldgB(const _Float16* __restrict__ w, int lane)
{
    v8h lo = *(const v8h*)(w + lane * 16);
    v8h hi = *(const v8h*)(w + lane * 16 + 8);
    return __builtin_shufflevector(lo, hi, 0,1,2,3,4,5,6,7,8,9,10,11,12,13,14,15);
}

// ---- dual M-tile GEMM: 32xK @ Kx112, B fragments batched per k-tile -------
__device__ __forceinline__ void gemm2(const _Float16* act, int pitch,
                                      const _Float16* __restrict__ w,
                                      int KT, int lane, v8f acc0[NT], v8f acc1[NT])
{
#pragma unroll
    for (int nt = 0; nt < NT; ++nt) {
        v8f z = {0,0,0,0,0,0,0,0};
        acc0[nt] = z; acc1[nt] = z;
    }
    for (int kt = 0; kt < KT; ++kt) {
        v16h b[NT];
#pragma unroll
        for (int nt = 0; nt < NT; ++nt)
            b[nt] = ldgB(w + (size_t)(kt * NT + nt) * 512, lane);
        v16h a0 = ldsA(act,              pitch, lane, kt * 32);
        v16h a1 = ldsA(act + 16 * pitch, pitch, lane, kt * 32);
#pragma unroll
        for (int nt = 0; nt < NT; ++nt) {
            acc0[nt] = __builtin_amdgcn_wmma_f32_16x16x32_f16(
                false, a0, false, b[nt], (short)0, acc0[nt], false, false);
            acc1[nt] = __builtin_amdgcn_wmma_f32_16x16x32_f16(
                false, a1, false, b[nt], (short)0, acc1[nt], false, false);
        }
    }
}

// ---- result writers (C/D layout: lane -> n=(lane&15)+nt*16, rows rb+v) ----
__device__ __forceinline__ void store_f16_act(v8f acc[NT], _Float16* dst, int pitch,
                                              int rowoff, int lane,
                                              const float* __restrict__ bias, int do_tanh)
{
    int n0 = lane & 15, rb = (lane >> 4) * 8;
#pragma unroll
    for (int nt = 0; nt < NT; ++nt) {
        int n = n0 + nt * 16;
        float bv = (n < HN) ? bias[n] : 0.f;
#pragma unroll
        for (int v = 0; v < 8; ++v) {
            float x = acc[nt][v] + bv;
            if (do_tanh) x = fast_tanh(x);
            dst[(rowoff + rb + v) * pitch + n] = (_Float16)((n < HN) ? x : 0.f);
        }
    }
}
__device__ __forceinline__ void store_vec(v8f acc[NT], _Float16* dst, int rowoff, int lane)
{
    int n0 = lane & 15, rb = (lane >> 4) * 8;
#pragma unroll
    for (int nt = 0; nt < NT; ++nt) {
        int n = n0 + nt * 16;
#pragma unroll
        for (int v = 0; v < 8; ++v)
            dst[(rowoff + rb + v) * 112 + n] = (_Float16)acc[nt][v];
    }
}
__device__ __forceinline__ void store_hidden(v8f acc[NT], float* oh, int b0, int rowoff,
                                             int lane, const float* __restrict__ bias)
{
    int n0 = lane & 15, rb = (lane >> 4) * 8;
#pragma unroll
    for (int nt = 0; nt < NT; ++nt) {
        int n = n0 + nt * 16;
        if (n < HN) {
            float bv = bias[n];
#pragma unroll
            for (int v = 0; v < 8; ++v)
                oh[(size_t)(b0 + rowoff + rb + v) * HN + n] = fast_tanh(acc[nt][v] + bv);
        }
    }
}
__device__ __forceinline__ void store_encout(v8f acc[NT], _Float16* sFii, _Float16* sFij,
                                             int e, int rowoff, int lane,
                                             const float* __restrict__ bias)
{
    int n0 = lane & 15, rb = (lane >> 4) * 8;
#pragma unroll
    for (int nt = 0; nt < NT; ++nt) {
        int n = n0 + nt * 16;
        float bv = (n < HN) ? bias[n] : 0.f;
#pragma unroll
        for (int v = 0; v < 8; ++v) {
            float x = acc[nt][v] + bv;
            int r = rowoff + rb + v;
            if (e == 0)      sFii[r * 128 + n] = (_Float16)((n < HN) ? x : 0.f);
            else if (n < HN) sFij[(e - 1) * 32 * HN + r * HN + n] = (_Float16)x;
        }
    }
}

struct Params {
    const float* share;
    const float* inp;
    const float* rnn;
    const int*   agent;
    const float* a1b;  const float* a2b;
    const float* selfW; const float* selfB;
    const float* agW;   const float* agB;
    const float* ldW;   const float* ldB;
    const float* efcB;  const float* eoutB;
    const float* fcB;   const float* cr1B;
    const float* w2;    const float* b2;
    const _Float16* ws;
    float* out;
    int Bsz;
};

__global__ __launch_bounds__(32) void att_fused(Params P)
{
    // 32 batch rows per wave (two 16-row M tiles); all buffers indexed by
    // global row r in [0,32).
    __shared__ __align__(16) _Float16 sAct[32 * AP];     // 24576 B
    __shared__ __align__(16) _Float16 sOut[32 * 128];    //  8192 B
    __shared__ __align__(16) _Float16 sFii[32 * 128];    //  8192 B
    __shared__ __align__(16) _Float16 sVec[32 * 112];    //  7168 B
    __shared__ __align__(16) _Float16 sFij[2 * 32 * HN]; // 12800 B
    __shared__ __align__(16) float    sObs[32 * 18];     //  2304 B

    const int lane = threadIdx.x;
    const int b0   = blockIdx.x * 32;
    const int row  = lane & 15;
    const int sel  = lane >> 4;     // 0/1 half-wave selector

    // zero the f16 buffers once (pad columns must stay zero)
    {
        v8h z = {};
        for (int i = lane * 8; i < 32 * AP;  i += 256) *(v8h*)(sAct + i) = z;
        for (int i = lane * 8; i < 32 * 128; i += 256) *(v8h*)(sOut + i) = z;
        for (int i = lane * 8; i < 32 * 128; i += 256) *(v8h*)(sFii + i) = z;
    }

    const int ai = P.agent[0];
    const int j0 = (ai == 0) ? 1 : 0;
    const int j1 = (ai == 2) ? 1 : 2;

    // ---- load self observations (32 rows) ----
    for (int idx = lane; idx < 32 * 18; idx += 32) {
        int r = idx / 18, c = idx % 18;
        sObs[idx] = P.inp[(size_t)(b0 + r) * 18 + c];
    }

    // ================= actor MLP =================
    for (int idx = lane; idx < 32 * 32; idx += 32) {
        int r = idx >> 5, k = idx & 31;
        sAct[r * AP + k] = (_Float16)((k < 18) ? sObs[r * 18 + k] : 0.f);
    }
    {
        v8f acc0[NT], acc1[NT];
        gemm2(sAct, AP, P.ws + OFF_A1, 1, lane, acc0, acc1);
        store_f16_act(acc0, sOut, 128, 0,  lane, P.a1b, 1);
        store_f16_act(acc1, sOut, 128, 16, lane, P.a1b, 1);
        gemm2(sOut, 128, P.ws + OFF_A2, 4, lane, acc0, acc1);
        float* oh = P.out + P.Bsz;  // hidden_actor block
        store_hidden(acc0, oh, b0, 0,  lane, P.a2b);
        store_hidden(acc1, oh, b0, 16, lane, P.a2b);
    }

    // ================= 3x obs_encode =================
    for (int e = 0; e < 3; ++e) {
        if (e > 0) {
            int jj = (e == 1) ? j0 : j1;
            for (int idx = lane; idx < 32 * 18; idx += 32) {
                int r = idx / 18, c = idx % 18;
                sObs[idx] = P.share[(size_t)(b0 + r) * 54 + jj * 18 + c];
            }
        }
        // self_emb -> segment 0 (k 0..99; 100..127 stay zero)
        for (int t = 0; t < 2; ++t) {
            int r = 16 * t + row;
            float x0 = sObs[r*18+0], x1 = sObs[r*18+1];
            float x2 = sObs[r*18+2], x3 = sObs[r*18+3];
            for (int h = sel; h < HN; h += 2) {
                float v = P.selfB[h] + x0*P.selfW[h] + x1*P.selfW[HN+h]
                        + x2*P.selfW[2*HN+h] + x3*P.selfW[3*HN+h];
                sAct[r * AP + h] = (_Float16)v;
            }
        }
        v8f acc0[NT], acc1[NT];

        // ---- agent attention ----
        gemm2(sAct, AP, P.ws + OFF_AGC, 4, lane, acc0, acc1);
        store_vec(acc0, sVec, 0,  lane);
        store_vec(acc1, sVec, 16, lane);
        for (int t = 0; t < 2; ++t) {
            int r = 16 * t + row;
            float xa = sObs[r*18 + 10 + 2*sel], ya = sObs[r*18 + 11 + 2*sel];
            float beta = 0.f;
            for (int h = 0; h < HN; ++h) {
                float emb = xa * P.agW[h] + ya * P.agW[HN + h] + P.agB[h];
                beta += (float)sVec[r * 112 + h] * emb;
            }
            float betaO = __shfl_xor(beta, 16, 32);
            float m = fmaxf(beta, betaO);
            float e0 = __expf(beta - m), e1 = __expf(betaO - m);
            float inv = 1.f / (e0 + e1);
            float aOwn = e0 * inv, aOth = e1 * inv;
            float al0 = sel ? aOth : aOwn;
            float al1 = sel ? aOwn : aOth;
            float xa0 = sObs[r*18+10], ya0 = sObs[r*18+11];
            float xa1 = sObs[r*18+12], ya1 = sObs[r*18+13];
            for (int h = sel; h < HN; h += 2) {
                float w0 = P.agW[h], w1 = P.agW[HN + h], bb = P.agB[h];
                float v = al0 * (xa0*w0 + ya0*w1 + bb) + al1 * (xa1*w0 + ya1*w1 + bb);
                sAct[r * AP + 128 + h] = (_Float16)v;
            }
            for (int h = HN + sel; h < 128; h += 2) sAct[r * AP + 128 + h] = (_Float16)0.f;
        }

        // ---- landmark attention (3 landmarks) ----
        gemm2(sAct, AP, P.ws + OFF_LDC, 4, lane, acc0, acc1);
        store_vec(acc0, sVec, 0,  lane);
        store_vec(acc1, sVec, 16, lane);
        for (int t = 0; t < 2; ++t) {
            int r = 16 * t + row;
            float xl = sObs[r*18 + 4 + 2*sel], yl = sObs[r*18 + 5 + 2*sel];
            float x2 = sObs[r*18 + 8],         y2 = sObs[r*18 + 9];
            float beta = 0.f, beta2 = 0.f;
            for (int h = 0; h < HN; ++h) {
                float w0 = P.ldW[h], w1 = P.ldW[HN + h], bb = P.ldB[h];
                float s = (float)sVec[r * 112 + h];
                beta  += s * (xl*w0 + yl*w1 + bb);
                beta2 += s * (x2*w0 + y2*w1 + bb);
            }
            float betaO = __shfl_xor(beta, 16, 32);
            float bl0 = sel ? betaO : beta;
            float bl1 = sel ? beta  : betaO;
            float bl2 = beta2;
            float m = fmaxf(fmaxf(bl0, bl1), bl2);
            float e0 = __expf(bl0 - m), e1 = __expf(bl1 - m), e2 = __expf(bl2 - m);
            float inv = 1.f / (e0 + e1 + e2);
            float al0 = e0*inv, al1 = e1*inv, al2 = e2*inv;
            float xl0 = sObs[r*18+4], yl0 = sObs[r*18+5];
            float xl1 = sObs[r*18+6], yl1 = sObs[r*18+7];
            for (int h = sel; h < HN; h += 2) {
                float w0 = P.ldW[h], w1 = P.ldW[HN + h], bb = P.ldB[h];
                float v = al0 * (xl0*w0 + yl0*w1 + bb)
                        + al1 * (xl1*w0 + yl1*w1 + bb)
                        + al2 * (x2 *w0 + y2 *w1 + bb);
                sAct[r * AP + 256 + h] = (_Float16)v;
            }
            for (int h = HN + sel; h < 128; h += 2) sAct[r * AP + 256 + h] = (_Float16)0.f;
        }

        // ---- gi = self_emb @ enc_fc_w (reads seg0, then overwrites seg0) ----
        gemm2(sAct, AP, P.ws + OFF_EFC, 4, lane, acc0, acc1);
        store_f16_act(acc0, sAct, AP, 0,  lane, P.efcB, 0);
        store_f16_act(acc1, sAct, AP, 16, lane, P.efcB, 0);

        // ---- f = concat(gi, v_a, v_l) @ enc_out_w ----
        gemm2(sAct, AP, P.ws + OFF_EOUT, 12, lane, acc0, acc1);
        store_encout(acc0, sFii, sFij, e, 0,  lane, P.eoutB);
        store_encout(acc1, sFii, sFij, e, 16, lane, P.eoutB);
    }

    // ================= critic head =================
    {
        v8f acc0[NT], acc1[NT];
        // s = f_ii @ corr_mat
        gemm2(sFii, 128, P.ws + OFF_CORR, 4, lane, acc0, acc1);
        store_vec(acc0, sVec, 0,  lane);
        store_vec(acc1, sVec, 16, lane);
        for (int t = 0; t < 2; ++t) {
            int r = 16 * t + row;
            float beta = 0.f;
            for (int h = 0; h < HN; ++h)
                beta += (float)sVec[r * 112 + h] * (float)sFij[sel * 32 * HN + r * HN + h];
            float betaO = __shfl_xor(beta, 16, 32);
            float m = fmaxf(beta, betaO);
            float e0 = __expf(beta - m), e1 = __expf(betaO - m);
            float inv = 1.f / (e0 + e1);
            float aOwn = e0 * inv, aOth = e1 * inv;
            float al0 = sel ? aOth : aOwn;
            float al1 = sel ? aOwn : aOth;
            for (int h = sel; h < HN; h += 2) {
                float v = al0 * (float)sFij[r * HN + h]
                        + al1 * (float)sFij[32 * HN + r * HN + h];
                sAct[r * AP + 128 + h] = (_Float16)v;
            }
            for (int h = HN + sel; h < 128; h += 2) sAct[r * AP + 128 + h] = (_Float16)0.f;
        }

        // gi2 = f_ii @ fc_w -> segment 0
        gemm2(sFii, 128, P.ws + OFF_FC, 4, lane, acc0, acc1);
        store_f16_act(acc0, sAct, AP, 0,  lane, P.fcB, 0);
        store_f16_act(acc1, sAct, AP, 16, lane, P.fcB, 0);

        // h = tanh(concat(gi2, vi) @ crit_w1 + b1)
        gemm2(sAct, AP, P.ws + OFF_CR1, 8, lane, acc0, acc1);
        store_f16_act(acc0, sOut, 128, 0,  lane, P.cr1B, 1);
        store_f16_act(acc1, sOut, 128, 16, lane, P.cr1B, 1);

        // value = h @ crit_w2 + b2
        for (int t = 0; t < 2; ++t) {
            int r = 16 * t + row;
            float part = 0.f;
            for (int h = sel; h < HN; h += 2) part += (float)sOut[r * 128 + h] * P.w2[h];
            part += __shfl_xor(part, 16, 32);
            if (sel == 0) P.out[b0 + r] = part + P.b2[0];
        }

        // rnn_hxs passthrough (32 rows, one per lane)
        P.out[(size_t)P.Bsz * 101 + b0 + lane] = P.rnn[b0 + lane];
    }
}

extern "C" void kernel_launch(void* const* d_in, const int* in_sizes, int n_in,
                              void* d_out, int out_size, void* d_ws, size_t ws_size,
                              hipStream_t stream)
{
    const float* share = (const float*)d_in[0];
    const float* inp   = (const float*)d_in[1];
    const int*   agent = (const int*)d_in[2];
    const float* rnn   = (const float*)d_in[3];
    const float* aw1 = (const float*)d_in[5];  const float* ab1 = (const float*)d_in[6];
    const float* aw2 = (const float*)d_in[7];  const float* ab2 = (const float*)d_in[8];
    const float* esw = (const float*)d_in[9];  const float* esb = (const float*)d_in[10];
    const float* eaw = (const float*)d_in[11]; const float* eab = (const float*)d_in[12];
    const float* elw = (const float*)d_in[13]; const float* elb = (const float*)d_in[14];
    const float* eac = (const float*)d_in[15]; const float* elc = (const float*)d_in[16];
    const float* efw = (const float*)d_in[17]; const float* efb = (const float*)d_in[18];
    const float* eow = (const float*)d_in[19]; const float* eob = (const float*)d_in[20];
    const float* cmat= (const float*)d_in[21];
    const float* fcw = (const float*)d_in[22]; const float* fcb = (const float*)d_in[23];
    const float* cw1 = (const float*)d_in[24]; const float* cb1 = (const float*)d_in[25];
    const float* cw2 = (const float*)d_in[26]; const float* cb2 = (const float*)d_in[27];

    int Bsz = in_sizes[1] / 18;
    _Float16* ws = (_Float16*)d_ws;

    struct PackJob { const float* W; int off, Kp, Korig, seg; };
    PackJob jobs[9] = {
        { aw1,  OFF_A1,   32, 18,  0 },
        { aw2,  OFF_A2,  128, 100, 0 },
        { eac,  OFF_AGC, 128, 100, 0 },
        { elc,  OFF_LDC, 128, 100, 0 },
        { efw,  OFF_EFC, 128, 100, 0 },
        { eow,  OFF_EOUT,384, 300, 1 },
        { cmat, OFF_CORR,128, 100, 0 },
        { fcw,  OFF_FC,  128, 100, 0 },
        { cw1,  OFF_CR1, 256, 200, 1 },
    };
    for (int i = 0; i < 9; ++i) {
        int total = jobs[i].Kp * 112;
        pack_w<<<(total + 255) / 256, 256, 0, stream>>>(
            jobs[i].W, ws + jobs[i].off, jobs[i].Kp, jobs[i].Korig, jobs[i].seg);
    }

    Params P;
    P.share = share; P.inp = inp; P.rnn = rnn; P.agent = agent;
    P.a1b = ab1; P.a2b = ab2;
    P.selfW = esw; P.selfB = esb;
    P.agW = eaw; P.agB = eab;
    P.ldW = elw; P.ldB = elb;
    P.efcB = efb; P.eoutB = eob;
    P.fcB = fcb; P.cr1B = cb1;
    P.w2 = cw2; P.b2 = cb2;
    P.ws = ws; P.out = (float*)d_out; P.Bsz = Bsz;

    att_fused<<<(Bsz + 31) / 32, 32, 0, stream>>>(P);
}
